// MessagePassingNN_20160576487823
// MI455X (gfx1250) — matrix operand order repaired and checked by reading, not compiled, not run on hardware
//
#include <hip/hip_runtime.h>

#define HID 128
#define NGRAPH 512
#define RUNITS 256
#define T_ITERS 8

typedef __attribute__((ext_vector_type(16))) __bf16 v16bf;
typedef __attribute__((ext_vector_type(8)))  __bf16 v8bf;
typedef __attribute__((ext_vector_type(8)))  float  v8f;

__device__ __forceinline__ unsigned short f2bf(float f) {
  union { float f; unsigned int u; } c; c.f = f;
  unsigned int u = c.u;
  unsigned int r = (u + 0x7FFFu + ((u >> 16) & 1u)) >> 16;  // RNE
  return (unsigned short)r;
}

__device__ __forceinline__ float seluf(float x) {
  const float sc = 1.0507009873554805f, al = 1.6732632423543772f;
  return x > 0.f ? sc * x : sc * al * (__expf(x) - 1.f);
}
__device__ __forceinline__ float sigmf(float x) { return 1.f / (1.f + __expf(-x)); }

__device__ __forceinline__ v16bf cat8(v8bf lo, v8bf hi) {
  return __builtin_shufflevector(lo, hi, 0,1,2,3,4,5,6,7,8,9,10,11,12,13,14,15);
}

// ---------------------------------------------------------------------------
// Repack a [kSteps*32, nCols] f32 weight matrix into WMMA B-fragment order:
// out[((nt*kSteps + ks)*32 + lane)*16 + j] = bf16(W[ks*32 + (lane>=16?16:0) + j][nt*16 + (lane&15)])
// (contiguous k order, same as the A loads: lanes 0-15 hold K=0..15, lanes 16-31 hold K=16..31)
// ---------------------------------------------------------------------------
__global__ void pack_b_kernel(const float* __restrict__ W, unsigned short* __restrict__ out,
                              int kSteps, int nCols, int total) {
  int id = blockIdx.x * blockDim.x + threadIdx.x;
  if (id >= total) return;
  int j  = id & 15;
  int l  = (id >> 4) & 31;
  int ks = (id >> 9) % kSteps;
  int nt = id / (512 * kSteps);
  int k = ks * 32 + ((l >> 4) & 1) * 16 + j;
  int n = nt * 16 + (l & 15);
  out[id] = f2bf(W[(long)k * nCols + n]);
}

__global__ void init_h_kernel(const float* __restrict__ f, float* __restrict__ h,
                              unsigned short* __restrict__ hbf, int total) {
  int i = blockIdx.x * blockDim.x + threadIdx.x;
  if (i < total) { float v = f[i]; h[i] = v; hbf[i] = f2bf(v); }
}

__global__ void zero_f32_kernel(float* __restrict__ p, int total) {
  int i = blockIdx.x * blockDim.x + threadIdx.x;
  if (i < total) p[i] = 0.f;
}

// ---------------------------------------------------------------------------
// Message kernel: 16 edges x 128 cols per block; 8 waves, wave w owns N-tile w.
// A = [h[first[e]] , h[second[e]]]  (16x256 bf16, staged in LDS)
// acc += A @ Wm ; msg = selu(acc + bm); agg[second[e]] += msg  (f32 atomics)
// ---------------------------------------------------------------------------
__global__ __launch_bounds__(256)
void msg_kernel(const unsigned short* __restrict__ hbf,
                const int* __restrict__ first, const int* __restrict__ second,
                const unsigned short* __restrict__ WmPk,
                const float* __restrict__ bm,
                float* __restrict__ agg, int P) {
  __shared__ alignas(32) unsigned short At[16 * 256];
  __shared__ int sDst[16];
  int t = threadIdx.x;
  int base = blockIdx.x * 16;

  // stage gathered/concatenated A tile (each thread: 16 bf16 = 32B)
  {
    int row = t >> 4, chunk = t & 15;
    int e = base + row; if (e >= P) e = P - 1;
    int c0 = chunk * 16;
    const unsigned short* src = (c0 < HID)
        ? (hbf + (long)first[e] * HID + c0)
        : (hbf + (long)second[e] * HID + (c0 - HID));
    const uint4* s4 = (const uint4*)src;
    uint4* d4 = (uint4*)&At[row * 256 + c0];
    d4[0] = s4[0]; d4[1] = s4[1];
    if (t < 16) { int ee = base + t; sDst[t] = (ee < P) ? second[ee] : -1; }
  }
  __syncthreads();

  int lane = t & 31, w = t >> 5;
  int sel8 = ((lane >> 4) & 1) * 16;
  v8f acc = {};
  const unsigned short* bpk = WmPk + ((long)(w * 8) * 32 + lane) * 16;
#pragma unroll
  for (int ks = 0; ks < 8; ++ks) {
    int off = (lane & 15) * 256 + ks * 32 + sel8;
    v8bf lo = *(const v8bf*)&At[off];
    v8bf hi = *(const v8bf*)&At[off + 8];
    v16bf a = cat8(lo, hi);
    v16bf b = *(const v16bf*)(bpk + (long)ks * 32 * 16);
    acc = __builtin_amdgcn_wmma_f32_16x16x32_bf16(false, a, false, b, (short)0, acc,
                                                  false, false);
  }
  int n = w * 16 + (lane & 15);
  float bias = bm[n];
  int mh = ((lane >> 4) & 1) * 8;
#pragma unroll
  for (int i = 0; i < 8; ++i) {
    int dst = sDst[i + mh];
    if (dst >= 0) {
      float v = seluf(acc[i] + bias);
      unsafeAtomicAdd(&agg[(long)dst * HID + n], v);
    }
  }
}

// ---------------------------------------------------------------------------
// Fused GRU: 16 nodes per block; wave w owns cols [16w,16w+16) of each gate.
// mx = agg@Wk + bk0 ; mh = h@Uk + bk1 ; gates z,r ; h' = z*h + (1-z)*tanh(xh + r*hh)
// ---------------------------------------------------------------------------
__global__ __launch_bounds__(256)
void gru_kernel(const float* __restrict__ agg,
                float* __restrict__ h, unsigned short* __restrict__ hbf,
                const unsigned short* __restrict__ WkPk,
                const unsigned short* __restrict__ UkPk,
                const float* __restrict__ bk, int N) {
  __shared__ alignas(32) unsigned short aT[16 * 128];
  __shared__ alignas(32) unsigned short hT[16 * 128];
  int t = threadIdx.x;
  int base = blockIdx.x * 16;

#pragma unroll
  for (int j = 0; j < 8; ++j) {
    int idx = t * 8 + j;
    int r = idx >> 7, c = idx & 127;
    int row = base + r; if (row >= N) row = N - 1;
    aT[idx] = f2bf(agg[(long)row * HID + c]);
    hT[idx] = hbf[(long)row * HID + c];
  }
  __syncthreads();

  int lane = t & 31, w = t >> 5;
  int sel8 = ((lane >> 4) & 1) * 16;
  v8f accZ = {}, accR = {}, accHX = {}, accHH = {};
#pragma unroll
  for (int ks = 0; ks < 4; ++ks) {
    int off = (lane & 15) * 128 + ks * 32 + sel8;
    v16bf aA = cat8(*(const v8bf*)&aT[off], *(const v8bf*)&aT[off + 8]);
    v16bf aH = cat8(*(const v8bf*)&hT[off], *(const v8bf*)&hT[off + 8]);
    v16bf wz = *(const v16bf*)(WkPk + ((long)(((w     ) * 4 + ks)) * 32 + lane) * 16);
    v16bf wr = *(const v16bf*)(WkPk + ((long)(((w +  8) * 4 + ks)) * 32 + lane) * 16);
    v16bf wh = *(const v16bf*)(WkPk + ((long)(((w + 16) * 4 + ks)) * 32 + lane) * 16);
    v16bf uz = *(const v16bf*)(UkPk + ((long)(((w     ) * 4 + ks)) * 32 + lane) * 16);
    v16bf ur = *(const v16bf*)(UkPk + ((long)(((w +  8) * 4 + ks)) * 32 + lane) * 16);
    v16bf uh = *(const v16bf*)(UkPk + ((long)(((w + 16) * 4 + ks)) * 32 + lane) * 16);
    accZ  = __builtin_amdgcn_wmma_f32_16x16x32_bf16(false, aA, false, wz, (short)0, accZ,  false, false);
    accZ  = __builtin_amdgcn_wmma_f32_16x16x32_bf16(false, aH, false, uz, (short)0, accZ,  false, false);
    accR  = __builtin_amdgcn_wmma_f32_16x16x32_bf16(false, aA, false, wr, (short)0, accR,  false, false);
    accR  = __builtin_amdgcn_wmma_f32_16x16x32_bf16(false, aH, false, ur, (short)0, accR,  false, false);
    accHX = __builtin_amdgcn_wmma_f32_16x16x32_bf16(false, aA, false, wh, (short)0, accHX, false, false);
    accHH = __builtin_amdgcn_wmma_f32_16x16x32_bf16(false, aH, false, uh, (short)0, accHH, false, false);
  }

  int n = w * 16 + (lane & 15);
  float b0z = bk[n],         b1z = bk[384 + n];
  float b0r = bk[128 + n],   b1r = bk[384 + 128 + n];
  float b0h = bk[256 + n],   b1h = bk[384 + 256 + n];
  int mh = ((lane >> 4) & 1) * 8;
#pragma unroll
  for (int i = 0; i < 8; ++i) {
    int row = base + i + mh;
    if (row < N) {
      float z  = sigmf(accZ[i] + b0z + b1z);
      float r  = sigmf(accR[i] + b0r + b1r);
      float hh = tanhf(accHX[i] + b0h + r * (accHH[i] + b1h));
      float ho = h[(long)row * HID + n];
      float hn = z * ho + (1.f - z) * hh;
      h[(long)row * HID + n] = hn;
      hbf[(long)row * HID + n] = f2bf(hn);
    }
  }
}

__global__ void pool_kernel(const float* __restrict__ h, const int* __restrict__ gid,
                            float* __restrict__ pooled, int N) {
  int node = blockIdx.x;
  int c = threadIdx.x;
  if (node < N) unsafeAtomicAdd(&pooled[(long)gid[node] * HID + c],
                                h[(long)node * HID + c]);
}

__global__ void mlp_dense_kernel(const float* __restrict__ X, const float* __restrict__ W,
                                 const float* __restrict__ b, float* __restrict__ Y,
                                 int rows, int inDim, int outDim, int applySelu) {
  int id = blockIdx.x * blockDim.x + threadIdx.x;
  if (id >= rows * outDim) return;
  int g = id / outDim, j = id % outDim;
  float acc = b[j];
  for (int c = 0; c < inDim; ++c) acc += X[(long)g * inDim + c] * W[(long)c * outDim + j];
  Y[id] = applySelu ? seluf(acc) : acc;
}

// ---------------------------------------------------------------------------
extern "C" void kernel_launch(void* const* d_in, const int* in_sizes, int n_in,
                              void* d_out, int out_size, void* d_ws, size_t ws_size,
                              hipStream_t stream) {
  const float* features = (const float*)d_in[0];
  const int*   topo     = (const int*)d_in[1];
  const int*   gids     = (const int*)d_in[2];
  const float* Wm = (const float*)d_in[3];
  const float* bm = (const float*)d_in[4];
  const float* Wk = (const float*)d_in[5];
  const float* Uk = (const float*)d_in[6];
  const float* bk = (const float*)d_in[7];
  const float* W1 = (const float*)d_in[8];
  const float* b1 = (const float*)d_in[9];
  const float* W2 = (const float*)d_in[10];
  const float* b2 = (const float*)d_in[11];
  const float* W3 = (const float*)d_in[12];
  const float* b3 = (const float*)d_in[13];

  int N = in_sizes[0] / HID;
  int P = in_sizes[1] / 2;
  const int* first  = topo;
  const int* second = topo + P;

  char* ws = (char*)d_ws;
  size_t off = 0;
  auto alloc = [&](size_t bytes) -> void* {
    void* p = ws + off;
    off += (bytes + 255) & ~(size_t)255;
    return p;
  };
  float*          h      = (float*)alloc((size_t)N * HID * 4);
  unsigned short* hbf    = (unsigned short*)alloc((size_t)N * HID * 2);
  float*          agg    = (float*)alloc((size_t)N * HID * 4);
  unsigned short* WmPk   = (unsigned short*)alloc((size_t)8 * 8 * 32 * 16 * 2);
  unsigned short* WkPk   = (unsigned short*)alloc((size_t)24 * 4 * 32 * 16 * 2);
  unsigned short* UkPk   = (unsigned short*)alloc((size_t)24 * 4 * 32 * 16 * 2);
  float*          pooled = (float*)alloc((size_t)NGRAPH * HID * 4);
  float*          x1     = (float*)alloc((size_t)NGRAPH * RUNITS * 4);
  float*          x2     = (float*)alloc((size_t)NGRAPH * RUNITS * 4);
  (void)ws_size; (void)n_in; (void)out_size;

  // weight repack + state init
  { int tot = 8 * 8 * 32 * 16;
    pack_b_kernel<<<(tot + 255) / 256, 256, 0, stream>>>(Wm, WmPk, 8, HID, tot); }
  { int tot = 24 * 4 * 32 * 16;
    pack_b_kernel<<<(tot + 255) / 256, 256, 0, stream>>>(Wk, WkPk, 4, 3 * HID, tot); }
  { int tot = 24 * 4 * 32 * 16;
    pack_b_kernel<<<(tot + 255) / 256, 256, 0, stream>>>(Uk, UkPk, 4, 3 * HID, tot); }
  { int tot = N * HID;
    init_h_kernel<<<(tot + 255) / 256, 256, 0, stream>>>(features, h, hbf, tot); }

  int mBlocks = (P + 15) / 16;
  int gBlocks = (N + 15) / 16;
  for (int it = 0; it < T_ITERS; ++it) {
    { int tot = N * HID;
      zero_f32_kernel<<<(tot + 255) / 256, 256, 0, stream>>>(agg, tot); }
    msg_kernel<<<mBlocks, 256, 0, stream>>>(hbf, first, second, WmPk, bm, agg, P);
    gru_kernel<<<gBlocks, 256, 0, stream>>>(agg, h, hbf, WkPk, UkPk, bk, N);
  }

  { int tot = NGRAPH * HID;
    zero_f32_kernel<<<(tot + 255) / 256, 256, 0, stream>>>(pooled, tot); }
  pool_kernel<<<N, HID, 0, stream>>>(h, gids, pooled, N);

  mlp_dense_kernel<<<(NGRAPH * RUNITS + 255) / 256, 256, 0, stream>>>(
      pooled, W1, b1, x1, NGRAPH, HID, RUNITS, 1);
  mlp_dense_kernel<<<(NGRAPH * RUNITS + 255) / 256, 256, 0, stream>>>(
      x1, W2, b2, x2, NGRAPH, RUNITS, RUNITS, 1);
  mlp_dense_kernel<<<(NGRAPH + 255) / 256, 256, 0, stream>>>(
      x2, W3, b3, (float*)d_out, NGRAPH, RUNITS, 1, 0);
}
